// GNNLayer_35716948034357
// MI455X (gfx1250) — matrix operand order repaired and checked by reading, not compiled, not run on hardware
//
#include <hip/hip_runtime.h>

#define N_NODES 100000
#define N_EDGES 640000
#define DIM     128

typedef __attribute__((ext_vector_type(16))) __bf16 v16bf;
typedef __attribute__((ext_vector_type(8)))  float  v8f;

#define AGG_BYTES  (51200000u)   // N_NODES*DIM*4, 64B aligned
#define WFRAG_SLOT 64            // bytes per lane-slot: 16 bf16 hi + 16 bf16 lo
#define AS_SLOT    80            // LDS slot stride: 64B data + 16B pad (bank spread)

// split f = hi(bf16, truncated) + lo(bf16 of residual); returns packed 16-bit halves
__device__ __forceinline__ void bf_split(float f, unsigned& hib, unsigned& lob) {
    unsigned u = __builtin_bit_cast(unsigned, f);
    hib = u >> 16;
    float hf = __builtin_bit_cast(float, u & 0xffff0000u);
    lob = __builtin_bit_cast(unsigned, f - hf) >> 16;
}

// ---------------- kernel 1: zero the aggregation buffer ----------------
__global__ void gnn_zero_kernel(float4* __restrict__ agg4) {
    int i = blockIdx.x * blockDim.x + threadIdx.x;
    agg4[i] = make_float4(0.f, 0.f, 0.f, 0.f);
}

// ---------------- kernel 2: edge scatter-add (segment_sum) ----------------
// one thread per (edge, feature); 81.9M f32 atomics resolving in 192MB L2
__global__ void gnn_scatter_kernel(const float* __restrict__ x,
                                   const long long* __restrict__ ei,
                                   float* __restrict__ agg) {
    long long t = (long long)blockIdx.x * blockDim.x + threadIdx.x;
    int e = (int)(t >> 7);
    int f = (int)(t & 127);
    int src = (int)ei[e];
    int dst = (int)ei[(long long)N_EDGES + e];
    atomicAdd(&agg[(long long)dst * DIM + f], x[(long long)src * DIM + f]);
}

// ---------------- kernel 3: one-time W -> bf16 hi/lo fragment layout ----------------
// B-fragment mapping used by this code (contiguous K per lane half, same as A):
//   lane: n = n0+lmod, half lhalf; element j -> K = kk*32 + lhalf*16 + j
// slot = ((wave*4 + kk)*2 + lhalf)*16 + lmod ; 64B/slot: [0,32) hi, [32,64) lo
__global__ void gnn_wprep_kernel(const float* __restrict__ W, // [K][N] row-major
                                 unsigned short* __restrict__ wf) {
    int t = blockIdx.x * blockDim.x + threadIdx.x;  // 16384 = 128*128
    int n = t & 127, K = t >> 7;
    int wave = n >> 4, lmod = n & 15;
    int kk = K >> 5, lhalf = (K >> 4) & 1, j = K & 15;
    int slot = ((wave * 4 + kk) * 2 + lhalf) * 16 + lmod;
    unsigned hib, lob;
    bf_split(W[K * DIM + n], hib, lob);
    wf[slot * (WFRAG_SLOT / 2) + j]      = (unsigned short)hib;
    wf[slot * (WFRAG_SLOT / 2) + 16 + j] = (unsigned short)lob;
}

// ---------------- kernel 4: fused (x+agg)@W + b, ReLU via bf16 WMMA ----------------
// block = 256 threads = 8 waves; block owns 16 rows, wave owns 16 cols.
// A tile pre-split into LDS in fragment order; hot loop = pure b128 loads + WMMA.
__global__ void __launch_bounds__(256)
gnn_gemm_kernel(const float* __restrict__ x,
                const float* __restrict__ agg,
                const char* __restrict__ wfrag,
                const float* __restrict__ bias,
                float* __restrict__ out) {
    // A fragments: slot = (kk*2 + lhalf)*16 + m, stride 80B: [0,32) hi, [32,64) lo
    __shared__ __align__(16) char As[4 * 2 * 16 * AS_SLOT];   // 10.0 KB

    const int m0  = blockIdx.x * 16;
    const int tid = threadIdx.x;

    // ---- staging: each thread owns one 8-element half-slot ----
    // A-fragment mapping: lane m, half h; element j -> K = kk*32 + h*16 + j
    // j in [0,8)  -> K = kk*32 + h*16     + j   (contiguous 8)
    // j in [8,16) -> K = kk*32 + h*16 + 8 + (j-8) (contiguous 8)
    {
        int m    = tid & 15;
        int half = (tid >> 4) & 1;          // which run of 8 contiguous K
        int h    = (tid >> 5) & 1;          // lhalf
        int kk   = tid >> 6;                // K chunk
        int K0   = kk * 32 + h * 16 + half * 8;
        long long g = ((long long)(m0 + m) * DIM + K0) >> 2;  // float4 index
        float4 xa = ((const float4*)x)[g],     xb = ((const float4*)x)[g + 1];
        float4 aa = ((const float4*)agg)[g],   ab = ((const float4*)agg)[g + 1];
        float s[8] = { xa.x + aa.x, xa.y + aa.y, xa.z + aa.z, xa.w + aa.w,
                       xb.x + ab.x, xb.y + ab.y, xb.z + ab.z, xb.w + ab.w };
        unsigned hp[4], lp[4];
        #pragma unroll
        for (int p = 0; p < 4; ++p) {
            unsigned h0, l0, h1, l1;
            bf_split(s[2 * p],     h0, l0);
            bf_split(s[2 * p + 1], h1, l1);
            hp[p] = h0 | (h1 << 16);
            lp[p] = l0 | (l1 << 16);
        }
        int slot = (kk * 2 + h) * 16 + m;
        char* base = As + slot * AS_SLOT + half * 16;
        *(int4*)(base)      = make_int4(hp[0], hp[1], hp[2], hp[3]);
        *(int4*)(base + 32) = make_int4(lp[0], lp[1], lp[2], lp[3]);
    }
    __syncthreads();

    const int lane  = tid & 31;
    const int wave  = tid >> 5;
    const int n0    = wave * 16;
    const int lhalf = lane >> 4;
    const int lmod  = lane & 15;

    v8f acc = {};

    #pragma unroll
    for (int kk = 0; kk < 4; ++kk) {
        const char* ab = As + ((kk * 2 + lhalf) * 16 + lmod) * AS_SLOT;
        v16bf ahi = *(const v16bf*)(ab);
        v16bf alo = *(const v16bf*)(ab + 32);

        const char* bb = wfrag +
            (((wave * 4 + kk) * 2 + lhalf) * 16 + lmod) * WFRAG_SLOT;
        v16bf bhi = *(const v16bf*)(bb);
        v16bf blo = *(const v16bf*)(bb + 32);

        // 3-term bf16 split product ~ fp32 accuracy; GEMM stays memory-bound
        acc = __builtin_amdgcn_wmma_f32_16x16x32_bf16(false, ahi, false, bhi,
                                                      (short)0, acc, false, false);
        acc = __builtin_amdgcn_wmma_f32_16x16x32_bf16(false, ahi, false, blo,
                                                      (short)0, acc, false, false);
        acc = __builtin_amdgcn_wmma_f32_16x16x32_bf16(false, alo, false, bhi,
                                                      (short)0, acc, false, false);
    }

    // ---- epilogue: bias + ReLU; D layout: VGPR v, lane L -> M = v + 8*(L>>4), N = L&15
    const float bv = bias[n0 + lmod];
    #pragma unroll
    for (int v = 0; v < 8; ++v) {
        int m = m0 + v + 8 * lhalf;
        float r = acc[v] + bv;
        out[(long long)m * DIM + n0 + lmod] = r > 0.f ? r : 0.f;
    }
}

extern "C" void kernel_launch(void* const* d_in, const int* in_sizes, int n_in,
                              void* d_out, int out_size, void* d_ws, size_t ws_size,
                              hipStream_t stream) {
    const float*     x  = (const float*)d_in[0];
    const long long* ei = (const long long*)d_in[1];
    const float*     W  = (const float*)d_in[2];
    const float*     b  = (const float*)d_in[3];
    float* out = (float*)d_out;

    float* agg   = (float*)d_ws;                       // 51.2 MB
    char*  wfrag = (char*)d_ws + AGG_BYTES;            // 64 KB fragment buffer

    // 1) zero agg: 3.2M float4 -> 12500 blocks
    gnn_zero_kernel<<<(N_NODES * DIM / 4) / 256, 256, 0, stream>>>((float4*)agg);

    // 2) one-time W fragment prep: 16384 threads
    gnn_wprep_kernel<<<(DIM * DIM) / 256, 256, 0, stream>>>(W, (unsigned short*)wfrag);

    // 3) scatter-add: 320000 blocks
    gnn_scatter_kernel<<<(N_EDGES * 128) / 256, 256, 0, stream>>>(x, ei, agg);

    // 4) fused WMMA GEMM + bias + ReLU: 6250 row tiles
    gnn_gemm_kernel<<<N_NODES / 16, 256, 0, stream>>>(x, agg, wfrag, b, out);
}
